// IPAttnProcessor2_0_cross_modal_9904194585118
// MI455X (gfx1250) — matrix operand order repaired and checked by reading, not compiled, not run on hardware
//
#include <hip/hip_runtime.h>

// ---------------------------------------------------------------------------
// Types for CDNA5 WMMA (wave32, 16x16x32 f16 -> f32)
// ---------------------------------------------------------------------------
typedef __attribute__((ext_vector_type(16))) _Float16 v16h;
typedef __attribute__((ext_vector_type(8)))  _Float16 v8h;
typedef __attribute__((ext_vector_type(4)))  _Float16 v4h;
typedef __attribute__((ext_vector_type(8)))  float    v8f;
typedef __attribute__((address_space(3)))    char     lds_char;

#define HEADS 20
#define HEAD_DIM 64
#define HIDDEN 1280
#define CROSS 2048
#define SEQ 4096
#define BATCH 4
#define KV_TEXT 77
#define KV_TEXT_PAD 96   // 3 k-steps of 32
#define KV_IP 4
#define KV_IP_PAD 32     // 1 k-step of 32

__device__ __forceinline__ v8f wmma16(v16h a, v16h b, v8f c) {
  return __builtin_amdgcn_wmma_f32_16x16x32_f16(false, a, false, b, (short)0, c,
                                                false, false);
}

// A fragment (16x32 f16, MxK): lane holds row M = lane%16.
// element j in [0,16) -> k = kbase + half*16 + j
__device__ __forceinline__ v16h load_a_frag(const _Float16* rowptr, int kbase, int half) {
  const _Float16* p = rowptr + kbase + half * 16;
  v8h lo = *(const v8h*)(p);
  v8h hi = *(const v8h*)(p + 8);
  v16h r;
#pragma unroll
  for (int i = 0; i < 8; i++) { r[i] = lo[i]; r[i + 8] = hi[i]; }
  return r;
}

// B fragment (32x16 f16, KxN) from BT stored (N x K) row-major:
// lane holds column n (= its BT row); element j -> k = kbase + half*16 + j
__device__ __forceinline__ v16h load_b_frag(const _Float16* colptr, int kbase, int half) {
  return *(const v16h*)(colptr + kbase + half * 16);
}

// ---------------------------------------------------------------------------
// CDNA5 async global->LDS (16B per lane), tracked by ASYNCcnt
// ---------------------------------------------------------------------------
__device__ __forceinline__ void async_copy16(const void* gsrc, void* ldst) {
  unsigned int loff = (unsigned int)(unsigned long long)(lds_char*)ldst;
  asm volatile("global_load_async_to_lds_b128 %0, %1, off"
               :: "v"(loff), "v"(gsrc)
               : "memory");
}
__device__ __forceinline__ void wait_async0() {
  asm volatile("s_wait_asynccnt 0x0" ::: "memory");
}
// Workgroup barrier that only drains LDS (DScnt), NOT global loadcnt:
// keeps A-prefetch global loads in flight across the k-step boundary.
__device__ __forceinline__ void barrier_lds_only() {
  asm volatile("s_wait_dscnt 0x0\n\t"
               "s_barrier_signal -1\n\t"
               "s_barrier_wait -1" ::: "memory");
}

// ---------------------------------------------------------------------------
// Prep kernels
// ---------------------------------------------------------------------------
__global__ void cvt4_f32_f16(const float* __restrict__ in, _Float16* __restrict__ out,
                             int n4) {
  int i = blockIdx.x * blockDim.x + threadIdx.x;
  if (i < n4) {
    float4 v = ((const float4*)in)[i];
    v4h o;
    o[0] = (_Float16)v.x; o[1] = (_Float16)v.y;
    o[2] = (_Float16)v.z; o[3] = (_Float16)v.w;
    *(v4h*)(out + 4ull * i) = o;
  }
}

// in: (K x N) f32 row-major  -> out: (N x K) f16 row-major
__global__ void __launch_bounds__(256) transpose_cvt(const float* __restrict__ in,
                                                     _Float16* __restrict__ out,
                                                     int K, int N) {
  __shared__ _Float16 tile[32][33];
  int bk = blockIdx.x * 32, bn = blockIdx.y * 32;
  int tx = threadIdx.x & 31, ty = threadIdx.x >> 5;
#pragma unroll
  for (int i = 0; i < 32; i += 8)
    tile[ty + i][tx] = (_Float16)in[(size_t)(bk + ty + i) * N + bn + tx];
  __syncthreads();
#pragma unroll
  for (int i = 0; i < 32; i += 8)
    out[(size_t)(bn + ty + i) * K + bk + tx] = tile[tx][ty + i];
}

// ---------------------------------------------------------------------------
// Block-tiled GEMM core: block = 256 threads (8 waves).
// Block tile: M=256 (wave w -> rows m0..m0+31), N=64, K-step 32.
// B tile (64x32) staged in LDS via async DMA, double buffered.
// Each wave: 2 A-frags (prefetched) x 4 B-frags -> 8 WMMA per k-step.
// Barrier discipline per iteration i (buffers p=i%2):
//   s_wait_asynccnt 0   -> this wave's DMA into buf[p] done
//   lds-only barrier    -> all waves' DMAs done AND all reads of buf[p^1] done
//   issue DMA -> buf[p^1], compute from buf[p]
// ---------------------------------------------------------------------------
#define GEMM_CORE(A_PTR, WT_PTR, KDIM, m0, n0, acc, bst)                         \
  {                                                                              \
    int srow = threadIdx.x >> 2, schk = threadIdx.x & 3;                         \
    const _Float16* bsrc = (WT_PTR) + (size_t)((n0) + srow) * (KDIM) + schk * 8; \
    async_copy16(bsrc, &bst[0][srow * 32 + schk * 8]);                           \
    const _Float16* ar0 = (A_PTR) + (size_t)((m0) + col) * (KDIM);               \
    const _Float16* ar1 = ar0 + (size_t)16 * (KDIM);                             \
    v16h a0 = load_a_frag(ar0, 0, half);                                         \
    v16h a1 = load_a_frag(ar1, 0, half);                                         \
    for (int k = 0; k < (KDIM); k += 32) {                                       \
      int p = (k >> 5) & 1;                                                      \
      wait_async0();                                                             \
      barrier_lds_only();                                                        \
      v16h na0 = a0, na1 = a1;                                                   \
      if (k + 32 < (KDIM)) {                                                     \
        async_copy16(bsrc + k + 32, &bst[p ^ 1][srow * 32 + schk * 8]);          \
        na0 = load_a_frag(ar0, k + 32, half);                                    \
        na1 = load_a_frag(ar1, k + 32, half);                                    \
      }                                                                          \
      const _Float16* bb = &bst[p][0];                                           \
      _Pragma("unroll")                                                          \
      for (int nt = 0; nt < 4; nt++) {                                           \
        v16h bf = *(const v16h*)(bb + (nt * 16 + col) * 32 + half * 16);         \
        acc[0][nt] = wmma16(a0, bf, acc[0][nt]);                                 \
        acc[1][nt] = wmma16(a1, bf, acc[1][nt]);                                 \
      }                                                                          \
      a0 = na0; a1 = na1;                                                        \
    }                                                                            \
  }

// Q projection: (16384 x 1280) @ WqT -> Qbuf (b,h,s,d) f16
__global__ void __launch_bounds__(256) qproj_kernel(const _Float16* __restrict__ A,
                                                    const _Float16* __restrict__ WT,
                                                    _Float16* __restrict__ Q) {
  __shared__ __align__(32) _Float16 bst[2][64 * 32];
  int lane = threadIdx.x & 31, w = threadIdx.x >> 5;
  int half = lane >> 4, col = lane & 15;
  int h = blockIdx.y;
  int m0 = blockIdx.x * 256 + w * 32;
  v8f acc[2][4] = {};
  GEMM_CORE(A, WT, HIDDEN, m0, h * 64, acc, bst);
  int b = m0 >> 12;            // /4096
  int s0 = m0 & 4095;
  _Float16* qp = Q + (((size_t)(b * HEADS + h)) * SEQ + s0) * HEAD_DIM;
#pragma unroll
  for (int rs = 0; rs < 2; rs++)
#pragma unroll
    for (int nt = 0; nt < 4; nt++)
#pragma unroll
      for (int r = 0; r < 8; r++)
        qp[(size_t)(rs * 16 + r + 8 * half) * HEAD_DIM + nt * 16 + col] =
            (_Float16)acc[rs][nt][r];
}

// Out projection: hs2(f16) @ WoutT + b_out + residual -> f32 out
__global__ void __launch_bounds__(256) outproj_kernel(const _Float16* __restrict__ A,
                                                      const _Float16* __restrict__ WT,
                                                      const float* __restrict__ bias,
                                                      const float* __restrict__ resid,
                                                      float* __restrict__ out) {
  __shared__ __align__(32) _Float16 bst[2][64 * 32];
  int lane = threadIdx.x & 31, w = threadIdx.x >> 5;
  int half = lane >> 4, col = lane & 15;
  int n0 = blockIdx.y * 64;
  int m0 = blockIdx.x * 256 + w * 32;
  v8f acc[2][4] = {};
  GEMM_CORE(A, WT, HIDDEN, m0, n0, acc, bst);
#pragma unroll
  for (int nt = 0; nt < 4; nt++) {
    int n = n0 + nt * 16 + col;
    float bv = bias[n];
#pragma unroll
    for (int rs = 0; rs < 2; rs++)
#pragma unroll
      for (int r = 0; r < 8; r++) {
        size_t row = (size_t)m0 + rs * 16 + r + 8 * half;
        out[row * HIDDEN + n] = acc[rs][nt][r] + bv + resid[row * HIDDEN + n];
      }
  }
}

// ---------------------------------------------------------------------------
// K / V projections from ctx f16 (4,81,2048).  (tiny: 324 rows total)
// transposed==0 : Out[b][h][key(kp)][64]   (key-major, for Q·K^T B-operand)
// transposed==1 : Out[b][h][64][key(kp)]   (d-major,  for P·V  B-operand)
// ---------------------------------------------------------------------------
__global__ void __launch_bounds__(32) kvproj_kernel(const _Float16* __restrict__ ctx,
                                                    const _Float16* __restrict__ WT,
                                                    _Float16* __restrict__ Out,
                                                    int key_off, int kv_valid, int nkt,
                                                    int transposed, int kp) {
  int lane = threadIdx.x & 31;
  int id = blockIdx.x;
  int h  = id % HEADS;
  int kt = (id / HEADS) % nkt;
  int b  = id / (HEADS * nkt);
  int half = lane >> 4, col = lane & 15;
  int keyrow = kt * 16 + col;
  if (keyrow >= kv_valid) keyrow = kv_valid - 1;     // clamp; stores are guarded
  const _Float16* arow = ctx + ((size_t)(b * 81) + key_off + keyrow) * CROSS;
  v8f acc[4] = {};
  for (int k = 0; k < CROSS; k += 32) {
    v16h af = load_a_frag(arow, k, half);
#pragma unroll
    for (int nt = 0; nt < 4; nt++) {
      const _Float16* bp = WT + (size_t)(h * 64 + nt * 16 + col) * CROSS;
      acc[nt] = wmma16(af, load_b_frag(bp, k, half), acc[nt]);
    }
  }
  if (!transposed) {
    _Float16* op = Out + (size_t)(b * HEADS + h) * kp * HEAD_DIM;
#pragma unroll
    for (int nt = 0; nt < 4; nt++)
#pragma unroll
      for (int r = 0; r < 8; r++) {
        int key = kt * 16 + r + 8 * half;
        if (key < kv_valid)
          op[(size_t)key * HEAD_DIM + nt * 16 + col] = (_Float16)acc[nt][r];
      }
  } else {
    _Float16* op = Out + (size_t)(b * HEADS + h) * HEAD_DIM * kp;
#pragma unroll
    for (int nt = 0; nt < 4; nt++) {
      int d = nt * 16 + col;
#pragma unroll
      for (int r = 0; r < 8; r++) {
        int key = kt * 16 + r + 8 * half;
        if (key < kv_valid) op[(size_t)d * kp + key] = (_Float16)acc[nt][r];
      }
    }
  }
}

// ---------------------------------------------------------------------------
// Attention core (per wave, 16 query rows). NKT = key tiles for scores,
// NKS = 32-wide k-steps for P·V. Adds wscale * softmax(QK^T/8) V into outAcc.
// ---------------------------------------------------------------------------
#define LDSLD 96
template <int NKT, int NKS>
__device__ __forceinline__ void attn_accum(const v16h (&qa)[2],
                                           const _Float16* __restrict__ Kmat,
                                           const _Float16* __restrict__ Vt, int vld,
                                           int kv_valid, _Float16* ldsP, int lane,
                                           v8f (&outAcc)[4], float wscale) {
  const float scale = 0.125f;  // 1/sqrt(64)
  int half = lane >> 4, col = lane & 15;
  v8f sc[NKT] = {};
#pragma unroll
  for (int t = 0; t < NKT; t++) {
    const _Float16* bp = Kmat + (size_t)(t * 16 + col) * HEAD_DIM;
    sc[t] = wmma16(qa[0], load_b_frag(bp, 0, half), sc[t]);
    sc[t] = wmma16(qa[1], load_b_frag(bp, 32, half), sc[t]);
  }
  float m[8];
#pragma unroll
  for (int r = 0; r < 8; r++) m[r] = -3.0e38f;
#pragma unroll
  for (int t = 0; t < NKT; t++) {
    bool valid = (t * 16 + col) < kv_valid;
#pragma unroll
    for (int r = 0; r < 8; r++) {
      float s = valid ? sc[t][r] * scale : -3.0e38f;
      sc[t][r] = s;
      m[r] = fmaxf(m[r], s);
    }
  }
#pragma unroll
  for (int r = 0; r < 8; r++) {
    m[r] = fmaxf(m[r], __shfl_xor(m[r], 1, 32));
    m[r] = fmaxf(m[r], __shfl_xor(m[r], 2, 32));
    m[r] = fmaxf(m[r], __shfl_xor(m[r], 4, 32));
    m[r] = fmaxf(m[r], __shfl_xor(m[r], 8, 32));
  }
  float sum[8] = {};
#pragma unroll
  for (int t = 0; t < NKT; t++)
#pragma unroll
    for (int r = 0; r < 8; r++) {
      float p = __expf(sc[t][r] - m[r]);   // masked -> exp(-huge) = 0
      sc[t][r] = p;
      sum[r] += p;
    }
#pragma unroll
  for (int r = 0; r < 8; r++) {
    sum[r] += __shfl_xor(sum[r], 1, 32);
    sum[r] += __shfl_xor(sum[r], 2, 32);
    sum[r] += __shfl_xor(sum[r], 4, 32);
    sum[r] += __shfl_xor(sum[r], 8, 32);
  }
  // stage P in LDS (C-layout -> A-layout round trip)
#pragma unroll
  for (int t = 0; t < NKT; t++)
#pragma unroll
    for (int r = 0; r < 8; r++)
      ldsP[(r + 8 * half) * LDSLD + t * 16 + col] = (_Float16)sc[t][r];
  for (int k = NKT * 16 + col; k < NKS * 32; k += 16)
#pragma unroll
    for (int r = 0; r < 8; r++) ldsP[(r + 8 * half) * LDSLD + k] = (_Float16)0.f;
  __syncthreads();
  v8f o[4] = {};
  const _Float16* arow = ldsP + col * LDSLD;
#pragma unroll
  for (int ks = 0; ks < NKS; ks++) {
    v16h pa = load_a_frag(arow, ks * 32, half);
#pragma unroll
    for (int nt = 0; nt < 4; nt++)
      o[nt] = wmma16(pa, load_b_frag(Vt + (size_t)(nt * 16 + col) * vld, ks * 32, half),
                     o[nt]);
  }
#pragma unroll
  for (int nt = 0; nt < 4; nt++)
#pragma unroll
    for (int r = 0; r < 8; r++) outAcc[nt][r] += o[nt][r] * (wscale / sum[r]);
  __syncthreads();  // LDS may be reused by the next phase
}

__global__ void __launch_bounds__(128) attn_kernel(const _Float16* __restrict__ Q,
                                                   const _Float16* __restrict__ K,
                                                   const _Float16* __restrict__ Vt,
                                                   const _Float16* __restrict__ Kip,
                                                   const _Float16* __restrict__ Vtip,
                                                   _Float16* __restrict__ hs2) {
  __shared__ __align__(32) _Float16 Plds[4][16 * LDSLD];
  int lane = threadIdx.x & 31;
  int w = threadIdx.x >> 5;
  int bid = blockIdx.x;
  int qb = bid & 63;
  int bh = bid >> 6;               // b*20 + h
  int b = bh / HEADS, h = bh % HEADS;
  int s0 = (qb * 4 + w) * 16;
  int half = lane >> 4, col = lane & 15;
  const _Float16* qrow = Q + ((size_t)bh * SEQ + s0 + col) * HEAD_DIM;
  v16h qa[2];
  qa[0] = load_a_frag(qrow, 0, half);
  qa[1] = load_a_frag(qrow, 32, half);
  v8f outAcc[4] = {};
  attn_accum<5, 3>(qa, K + (size_t)bh * KV_TEXT_PAD * HEAD_DIM,
                   Vt + (size_t)bh * HEAD_DIM * KV_TEXT_PAD, KV_TEXT_PAD, KV_TEXT,
                   &Plds[w][0], lane, outAcc, 1.0f);
  if (b & 1) {  // IP branch only for batches 1,3 (uniform per block)
    attn_accum<1, 1>(qa, Kip + (size_t)bh * 16 * HEAD_DIM,
                     Vtip + (size_t)bh * HEAD_DIM * KV_IP_PAD, KV_IP_PAD, KV_IP,
                     &Plds[w][0], lane, outAcc, 1.0f /* IP_SCALE */);
  }
  _Float16* op = hs2 + ((size_t)(b * SEQ + s0)) * HIDDEN + h * 64;
#pragma unroll
  for (int nt = 0; nt < 4; nt++)
#pragma unroll
    for (int r = 0; r < 8; r++)
      op[(size_t)(r + 8 * half) * HIDDEN + nt * 16 + col] = (_Float16)outAcc[nt][r];
}

// ---------------------------------------------------------------------------
// Host launcher
// ---------------------------------------------------------------------------
extern "C" void kernel_launch(void* const* d_in, const int* in_sizes, int n_in,
                              void* d_out, int out_size, void* d_ws, size_t ws_size,
                              hipStream_t stream) {
  (void)in_sizes; (void)n_in; (void)out_size; (void)ws_size;
  const float* hs    = (const float*)d_in[0];
  const float* enc   = (const float*)d_in[1];
  const float* Wq    = (const float*)d_in[2];
  const float* Wk    = (const float*)d_in[3];
  const float* Wv    = (const float*)d_in[4];
  const float* Wkip  = (const float*)d_in[5];
  const float* Wvip  = (const float*)d_in[6];
  const float* Wout  = (const float*)d_in[7];
  const float* bout  = (const float*)d_in[8];
  float* out = (float*)d_out;

  const size_t HS_ELEMS  = (size_t)BATCH * SEQ * HIDDEN;   // 20,971,520
  const size_t CTX_ELEMS = (size_t)BATCH * 81 * CROSS;     // 663,552

  char* ws = (char*)d_ws;
  size_t off = 0;
  auto carve = [&](size_t bytes) -> char* {
    char* p = ws + off;
    off += (bytes + 255) & ~(size_t)255;
    return p;
  };
  _Float16* hsb   = (_Float16*)carve(HS_ELEMS * 2);          // later reused as hs2
  _Float16* ctxb  = (_Float16*)carve(CTX_ELEMS * 2);
  _Float16* WqT   = (_Float16*)carve((size_t)HIDDEN * HIDDEN * 2);
  _Float16* WkT   = (_Float16*)carve((size_t)HIDDEN * CROSS * 2);
  _Float16* WvT   = (_Float16*)carve((size_t)HIDDEN * CROSS * 2);
  _Float16* WkipT = (_Float16*)carve((size_t)HIDDEN * CROSS * 2);
  _Float16* WvipT = (_Float16*)carve((size_t)HIDDEN * CROSS * 2);
  _Float16* WoutT = (_Float16*)carve((size_t)HIDDEN * HIDDEN * 2);
  _Float16* Qb    = (_Float16*)carve(HS_ELEMS * 2);
  char* kv0 = ws + off;
  _Float16* Kb   = (_Float16*)carve((size_t)BATCH * HEADS * KV_TEXT_PAD * HEAD_DIM * 2);
  _Float16* Vtb  = (_Float16*)carve((size_t)BATCH * HEADS * HEAD_DIM * KV_TEXT_PAD * 2);
  _Float16* Kipb = (_Float16*)carve((size_t)BATCH * HEADS * 16 * HEAD_DIM * 2);
  _Float16* Vtipb= (_Float16*)carve((size_t)BATCH * HEADS * HEAD_DIM * KV_IP_PAD * 2);
  size_t kvbytes = (size_t)((ws + off) - kv0);

  // 1) f32 -> f16 conversions
  cvt4_f32_f16<<<(int)((HS_ELEMS / 4 + 255) / 256), 256, 0, stream>>>(hs, hsb,
                                                                     (int)(HS_ELEMS / 4));
  cvt4_f32_f16<<<(int)((CTX_ELEMS / 4 + 255) / 256), 256, 0, stream>>>(
      enc, ctxb, (int)(CTX_ELEMS / 4));
  // 2) weight transposes (K x N) -> (N x K) f16
  transpose_cvt<<<dim3(HIDDEN / 32, HIDDEN / 32), 256, 0, stream>>>(Wq, WqT, HIDDEN, HIDDEN);
  transpose_cvt<<<dim3(CROSS / 32, HIDDEN / 32), 256, 0, stream>>>(Wk, WkT, CROSS, HIDDEN);
  transpose_cvt<<<dim3(CROSS / 32, HIDDEN / 32), 256, 0, stream>>>(Wv, WvT, CROSS, HIDDEN);
  transpose_cvt<<<dim3(CROSS / 32, HIDDEN / 32), 256, 0, stream>>>(Wkip, WkipT, CROSS, HIDDEN);
  transpose_cvt<<<dim3(CROSS / 32, HIDDEN / 32), 256, 0, stream>>>(Wvip, WvipT, CROSS, HIDDEN);
  transpose_cvt<<<dim3(HIDDEN / 32, HIDDEN / 32), 256, 0, stream>>>(Wout, WoutT, HIDDEN, HIDDEN);
  // 3) zero K/V pads
  hipMemsetAsync(kv0, 0, kvbytes, stream);
  // 4) projections
  qproj_kernel<<<dim3(SEQ * BATCH / 256, HEADS), 256, 0, stream>>>(hsb, WqT, Qb);
  kvproj_kernel<<<BATCH * 5 * HEADS, 32, 0, stream>>>(ctxb, WkT, Kb, 0, KV_TEXT, 5, 0,
                                                      KV_TEXT_PAD);
  kvproj_kernel<<<BATCH * 5 * HEADS, 32, 0, stream>>>(ctxb, WvT, Vtb, 0, KV_TEXT, 5, 1,
                                                      KV_TEXT_PAD);
  kvproj_kernel<<<BATCH * 1 * HEADS, 32, 0, stream>>>(ctxb, WkipT, Kipb, KV_TEXT, KV_IP,
                                                      1, 0, 16);
  kvproj_kernel<<<BATCH * 1 * HEADS, 32, 0, stream>>>(ctxb, WvipT, Vtipb, KV_TEXT, KV_IP,
                                                      1, 1, KV_IP_PAD);
  // 5) attention (text + conditional IP), writes f16 merged heads into hsb (reuse)
  attn_kernel<<<BATCH * HEADS * 64, 128, 0, stream>>>(Qb, Kb, Vtb, Kipb, Vtipb, hsb);
  // 6) out projection + bias + residual -> f32 output
  outproj_kernel<<<dim3(SEQ * BATCH / 256, HEADS), 256, 0, stream>>>(hsb, WoutT, bout,
                                                                     hs, out);
}
